// CrossAttention_64742337020465
// MI455X (gfx1250) — matrix operand order repaired and checked by reading, compiled, not run on hardware
//
#include <hip/hip_runtime.h>
#include <hip/hip_bf16.h>
#include <stdint.h>

// ---------------------------------------------------------------------------
// Problem constants
// ---------------------------------------------------------------------------
#define B_ROWS 32768
#define DMODEL 1024
static constexpr long BD = (long)B_ROWS * DMODEL;   // 33,554,432 elements
static constexpr long DD = (long)DMODEL * DMODEL;   // 1,048,576  elements

// ---------------------------------------------------------------------------
// Types
// ---------------------------------------------------------------------------
typedef __bf16 bf16_t;
typedef __bf16 bf16x8  __attribute__((ext_vector_type(8)));
typedef __bf16 bf16x16 __attribute__((ext_vector_type(16)));
typedef float  f32x8   __attribute__((ext_vector_type(8)));

__device__ __forceinline__ bf16_t f2bf(float f) {
    uint32_t u; __builtin_memcpy(&u, &f, 4);
    uint32_t r = u + 0x7FFFu + ((u >> 16) & 1u);   // round-to-nearest-even
    uint16_t s = (uint16_t)(r >> 16);
    bf16_t b; __builtin_memcpy(&b, &s, 2);
    return b;
}
__device__ __forceinline__ float bf2f(bf16_t b) {
    uint16_t s; __builtin_memcpy(&s, &b, 2);
    uint32_t u = ((uint32_t)s) << 16;
    float f; __builtin_memcpy(&f, &u, 4);
    return f;
}

// Async global -> LDS copy (CDNA5, tracked by ASYNCcnt). Per-lane: 16 bytes
// from the 64-bit global address into LDS at the 32-bit LDS byte offset.
// Per ISA 10.2: the low 32 bits of a flat LDS pointer ARE the LDS address.
__device__ __forceinline__ void async_load_b128(uint32_t lds_off, uint64_t gaddr) {
    asm volatile("global_load_async_to_lds_b128 %0, %1, off"
                 :: "v"(lds_off), "v"(gaddr) : "memory");
}
__device__ __forceinline__ void wait_asynccnt_le2() {
    asm volatile("s_wait_asynccnt 2" ::: "memory");
}
__device__ __forceinline__ void wait_asynccnt_0() {
    asm volatile("s_wait_asynccnt 0" ::: "memory");
}

// ---------------------------------------------------------------------------
// Kernel 1: f32 -> bf16 conversion of the three feature streams
//           (contiguous [3][B][D] bf16 output)
// ---------------------------------------------------------------------------
__global__ void k_convert_feats(const float* __restrict__ f1,
                                const float* __restrict__ f2,
                                const float* __restrict__ f3,
                                bf16_t* __restrict__ dst) {
    long idx  = (long)blockIdx.x * blockDim.x + threadIdx.x;
    long base = idx * 4;                       // 4 floats per thread
    if (base >= 3L * BD) return;
    int  s   = (int)(base / BD);               // BD % 4 == 0: no straddle
    long off = base - (long)s * BD;
    const float* src = (s == 0) ? f1 : ((s == 1) ? f2 : f3);
    const float4 v = *(const float4*)(src + off);
    bf16_t o[4] = { f2bf(v.x), f2bf(v.y), f2bf(v.z), f2bf(v.w) };
    uint64_t pk; __builtin_memcpy(&pk, o, 8);
    *(uint64_t*)(dst + base) = pk;             // b64 store
}

// ---------------------------------------------------------------------------
// Kernel 2: convert + transpose the 6 needed weight matrices to bf16 [N][K].
//   z: 0=Wq[2]  1=Wk[0]  2=Wk[1]  3=Wv[0]  4=Wv[1]  5=Wo
// ---------------------------------------------------------------------------
__global__ void k_transpose_w(const float* __restrict__ Wq,
                              const float* __restrict__ Wk,
                              const float* __restrict__ Wv,
                              const float* __restrict__ Wo,
                              bf16_t* __restrict__ wt) {
    __shared__ float tile[32][33];
    const float* src;
    switch (blockIdx.z) {
        case 0:  src = Wq + 2 * DD; break;
        case 1:  src = Wk;          break;
        case 2:  src = Wk + DD;     break;
        case 3:  src = Wv;          break;
        case 4:  src = Wv + DD;     break;
        default: src = Wo;          break;
    }
    bf16_t* dst = wt + (long)blockIdx.z * DD;
    const int tx = threadIdx.x, ty = threadIdx.y;      // block (32, 8)
    const int x = blockIdx.x * 32 + tx;                // e (src col)
    const int y = blockIdx.y * 32 + ty;                // d (src row)
    #pragma unroll
    for (int j = 0; j < 32; j += 8)
        tile[ty + j][tx] = src[(long)(y + j) * DMODEL + x];
    __syncthreads();
    const int x2 = blockIdx.y * 32 + tx;               // d (dst col)
    const int y2 = blockIdx.x * 32 + ty;               // e (dst row)
    #pragma unroll
    for (int j = 0; j < 32; j += 8)
        dst[(long)(y2 + j) * DMODEL + x2] = f2bf(tile[tx][ty + j]);
}

// ---------------------------------------------------------------------------
// Kernel 3: bf16 WMMA GEMM   out[M x 1024] = A[M x 1024] * Bt^T + bias
//   A  : row-major [M][1024] bf16  (direct global b128 fragment loads)
//   Bt : row-major [1024(N)][1024(K)] bf16 (pre-transposed weights),
//        double-buffered 128x32 tiles staged with GLOBAL_LOAD_ASYNC_TO_LDS_B128
//        (ASYNCcnt) and read back as WMMA fragments via ds_load_b128.
//   256 threads = 8 waves; block tile 128x128; wave tile 32(M) x 64(N);
//   K-step 32 -> 8 x v_wmma_f32_16x16x32_bf16 per wave per step.
// ---------------------------------------------------------------------------
template <bool STORE_F32>
__global__ void __launch_bounds__(256)
k_gemm_bf16(const bf16_t* __restrict__ A,
            const bf16_t* __restrict__ Bt,
            const float*  __restrict__ bias,
            bf16_t* __restrict__ outBf,
            float*  __restrict__ outF) {
    // Two 8KB B-tiles: 128 rows (N) x 32 elements (K) each, rows 64B apart.
    __shared__ __attribute__((aligned(64))) bf16_t sB[2][128 * 32];

    const int tid  = threadIdx.x;
    const int lane = tid & 31;
    const int w    = tid >> 5;               // 0..7
    const int wm   = w >> 1;                 // 0..3  -> 4 x 32 = 128 M
    const int wn   = w & 1;                  // 0..1  -> 2 x 64 = 128 N
    const int m0    = blockIdx.y * 128 + wm * 32;
    const int n0blk = blockIdx.x * 128;
    const int mrow = lane & 15;              // A-fragment row (M)
    const int half = lane >> 4;              // lane half selects K sub-block
    const int ncol = lane & 15;              // B-fragment column (N)

    // --- async staging addressing: thread t moves 32 contiguous bytes/step ---
    const int srow  = tid >> 1;              // 0..127 : local N row
    const int spart = (tid & 1) * 16;        // element offset inside 32-elem row
    const bf16_t* gB = Bt + (long)(n0blk + srow) * DMODEL + spart;
    uint32_t ldsDst[2];
    ldsDst[0] = (uint32_t)(uintptr_t)&sB[0][srow * 32 + spart];
    ldsDst[1] = (uint32_t)(uintptr_t)&sB[1][srow * 32 + spart];

    f32x8 vzero = {};
    f32x8 acc[2][4];
    #pragma unroll
    for (int mt = 0; mt < 2; ++mt)
        #pragma unroll
        for (int nt = 0; nt < 4; ++nt) acc[mt][nt] = vzero;

    // Per-ISA 16-bit A layout: lane L holds row m=L%16, K = half*8+{0..7}
    // in VGPR0..3 and K = 16+half*8+{0..7} in VGPR4..7.
    const bf16_t* aBase[2];
    #pragma unroll
    for (int mt = 0; mt < 2; ++mt)
        aBase[mt] = A + (long)(m0 + mt * 16 + mrow) * DMODEL;

    // Per-ISA 16-bit B layout: lane L holds column n=L%16,
    // K = half*8+{0..7} in elems 0..7 and K = 16+half*8+{0..7} in elems 8..15.
    const int bRowOff[4] = {
        (wn * 64 +  0 + ncol) * 32 + half * 8,
        (wn * 64 + 16 + ncol) * 32 + half * 8,
        (wn * 64 + 32 + ncol) * 32 + half * 8,
        (wn * 64 + 48 + ncol) * 32 + half * 8
    };

    // Prologue: stage K-slab 0 into buffer 0.
    async_load_b128(ldsDst[0],      (uint64_t)(uintptr_t)(gB));
    async_load_b128(ldsDst[0] + 16, (uint64_t)(uintptr_t)(gB + 8));

    for (int k0 = 0; k0 < DMODEL; k0 += 32) {
        const int cur = (k0 >> 5) & 1;
        if (k0 + 32 < DMODEL) {
            // stage next K-slab into the other buffer (read last iter,
            // release guaranteed by the trailing barrier of the previous iter)
            const bf16_t* gn = gB + k0 + 32;
            async_load_b128(ldsDst[cur ^ 1],      (uint64_t)(uintptr_t)(gn));
            async_load_b128(ldsDst[cur ^ 1] + 16, (uint64_t)(uintptr_t)(gn + 8));
            wait_asynccnt_le2();   // in-order: <=2 outstanding => cur complete
            __builtin_prefetch(aBase[0] + k0 + 32, 0, 1);
        } else {
            wait_asynccnt_0();
        }
        __syncthreads();           // cur tile visible to all waves

        bf16x16 afr[2];
        #pragma unroll
        for (int mt = 0; mt < 2; ++mt) {
            bf16x8 lo = *(const bf16x8*)(aBase[mt] + k0 + half * 8);
            bf16x8 hi = *(const bf16x8*)(aBase[mt] + k0 + 16 + half * 8);
            #pragma unroll
            for (int i = 0; i < 8; ++i) { afr[mt][i] = lo[i]; afr[mt][8 + i] = hi[i]; }
        }
        bf16x16 bfr[4];
        #pragma unroll
        for (int nt = 0; nt < 4; ++nt) {
            bf16x8 lo = *(const bf16x8*)&sB[cur][bRowOff[nt]];      // ds_load_b128
            bf16x8 hi = *(const bf16x8*)&sB[cur][bRowOff[nt] + 16]; // ds_load_b128
            #pragma unroll
            for (int i = 0; i < 8; ++i) { bfr[nt][i] = lo[i]; bfr[nt][8 + i] = hi[i]; }
        }
        #pragma unroll
        for (int mt = 0; mt < 2; ++mt)
            #pragma unroll
            for (int nt = 0; nt < 4; ++nt)
                acc[mt][nt] = __builtin_amdgcn_wmma_f32_16x16x32_bf16(
                    false, afr[mt], false, bfr[nt],
                    (short)0, acc[mt][nt], false, false);

        __syncthreads();           // all waves done reading cur before reuse
    }

    // Epilogue: C/D layout -> element r of v8f is row (half*8 + r), col = ncol.
    #pragma unroll
    for (int nt = 0; nt < 4; ++nt) {
        const int   col = n0blk + wn * 64 + nt * 16 + ncol;
        const float bv  = bias[col];
        #pragma unroll
        for (int mt = 0; mt < 2; ++mt) {
            #pragma unroll
            for (int r = 0; r < 8; ++r) {
                const long row = m0 + mt * 16 + half * 8 + r;
                const float v  = acc[mt][nt][r] + bv;
                if constexpr (STORE_F32) outF[row * DMODEL + col] = v;
                else                     outBf[row * DMODEL + col] = f2bf(v);
            }
        }
    }
}

// ---------------------------------------------------------------------------
// Kernel 4: stream-2 cross attention (the only non-degenerate softmax).
//   One wave32 per row: s_c = dot(q2, k_c)/sqrt(D), 2-way softmax, blend v0/v1.
// ---------------------------------------------------------------------------
__global__ void __launch_bounds__(256)
k_attend2(const bf16_t* __restrict__ q2, const bf16_t* __restrict__ k0p,
          const bf16_t* __restrict__ k1p, const bf16_t* __restrict__ v0p,
          const bf16_t* __restrict__ v1p, bf16_t* __restrict__ att) {
    const int  lane = threadIdx.x & 31;
    const int  w    = threadIdx.x >> 5;
    const long row  = (long)blockIdx.x * 8 + w;
    const long base = row * DMODEL;

    float s0 = 0.f, s1 = 0.f;
    #pragma unroll
    for (int it = 0; it < DMODEL / 256; ++it) {       // 4 iters x 8 elems/lane
        const int d  = it * 256 + lane * 8;
        bf16x8 qv = *(const bf16x8*)(q2  + base + d);
        bf16x8 k0 = *(const bf16x8*)(k0p + base + d);
        bf16x8 k1 = *(const bf16x8*)(k1p + base + d);
        #pragma unroll
        for (int e = 0; e < 8; ++e) {
            const float q = bf2f(qv[e]);
            s0 += q * bf2f(k0[e]);
            s1 += q * bf2f(k1[e]);
        }
    }
    #pragma unroll
    for (int off = 16; off >= 1; off >>= 1) {
        s0 += __shfl_xor(s0, off, 32);
        s1 += __shfl_xor(s1, off, 32);
    }
    const float scale = 0.03125f;                      // 1024^-0.5
    s0 *= scale; s1 *= scale;
    const float m   = fmaxf(s0, s1);
    const float e0  = __expf(s0 - m);
    const float e1  = __expf(s1 - m);
    const float inv = 1.f / (e0 + e1);
    const float p0  = e0 * inv, p1 = e1 * inv;

    #pragma unroll
    for (int it = 0; it < DMODEL / 256; ++it) {
        const int d = it * 256 + lane * 8;
        bf16x8 v0 = *(const bf16x8*)(v0p + base + d);
        bf16x8 v1 = *(const bf16x8*)(v1p + base + d);
        bf16x8 o;
        #pragma unroll
        for (int e = 0; e < 8; ++e)
            o[e] = f2bf(p0 * bf2f(v0[e]) + p1 * bf2f(v1[e]));
        *(bf16x8*)(att + base + d) = o;
    }
}

// ---------------------------------------------------------------------------
// Host-side orchestration
// ---------------------------------------------------------------------------
extern "C" void kernel_launch(void* const* d_in, const int* in_sizes, int n_in,
                              void* d_out, int out_size, void* d_ws, size_t ws_size,
                              hipStream_t stream) {
    (void)in_sizes; (void)n_in; (void)out_size; (void)ws_size;

    const float* f1 = (const float*)d_in[0];
    const float* f2 = (const float*)d_in[1];
    const float* f3 = (const float*)d_in[2];
    const float* Wq = (const float*)d_in[3];
    const float* bq = (const float*)d_in[4];
    const float* Wk = (const float*)d_in[5];
    const float* bk = (const float*)d_in[6];
    const float* Wv = (const float*)d_in[7];
    const float* bv = (const float*)d_in[8];
    const float* Wo = (const float*)d_in[9];
    const float* bo = (const float*)d_in[10];
    float* out = (float*)d_out;

    // Workspace carve-up (256B aligned regions)
    char*  ws  = (char*)d_ws;
    size_t off = 0;
    auto alloc = [&](size_t bytes) -> char* {
        char* p = ws + off;
        off += (bytes + 255) & ~(size_t)255;
        return p;
    };
    bf16_t* featsBf = (bf16_t*)alloc(3L * BD * sizeof(bf16_t)); // [3][B][D]
    bf16_t* wt      = (bf16_t*)alloc(6L * DD * sizeof(bf16_t)); // 6 transposed mats
    bf16_t* qkv     = (bf16_t*)alloc(5L * BD * sizeof(bf16_t)); // q2,k0,k1,v0,v1
    bf16_t* att2    = (bf16_t*)alloc(1L * BD * sizeof(bf16_t)); // attended[2]

    // 1) feature conversion (3*BD/4 threads, 4 f32 each)
    {
        long nThreads = 3L * BD / 4;
        dim3 g((unsigned)((nThreads + 255) / 256));
        k_convert_feats<<<g, 256, 0, stream>>>(f1, f2, f3, featsBf);
    }

    // 2) weight convert+transpose (only the 6 live matrices)
    k_transpose_w<<<dim3(32, 32, 6), dim3(32, 8), 0, stream>>>(Wq, Wk, Wv, Wo, wt);

    // 3) projection GEMMs: q2 = f3*Wq2, k0 = f1*Wk0, k1 = f2*Wk1,
    //                      v0 = f1*Wv0, v1 = f2*Wv1   (bias fused)
    const dim3 gg(DMODEL / 128, B_ROWS / 128);         // (8, 256)
    const bf16_t* Asrc[5]   = { featsBf + 2 * BD, featsBf, featsBf + BD,
                                featsBf,          featsBf + BD };
    const float*  biasSrc[5] = { bq + 2 * DMODEL, bk, bk + DMODEL,
                                 bv,              bv + DMODEL };
    for (int t = 0; t < 5; ++t)
        k_gemm_bf16<false><<<gg, 256, 0, stream>>>(
            Asrc[t], wt + (long)t * DD, biasSrc[t], qkv + (long)t * BD, nullptr);

    // 4) stream-2 attention (streams 0/1 have single-element softmax == 1)
    k_attend2<<<dim3(B_ROWS / 8), 256, 0, stream>>>(
        qkv, qkv + BD, qkv + 2 * BD, qkv + 3 * BD, qkv + 4 * BD, att2);

    // 5) output GEMMs: out0 = v1*Wo+bo, out1 = v0*Wo+bo, out2 = att2*Wo+bo
    const bf16_t* Ofin[3] = { qkv + 4 * BD, qkv + 3 * BD, att2 };
    for (int t = 0; t < 3; ++t)
        k_gemm_bf16<true><<<gg, 256, 0, stream>>>(
            Ofin[t], wt + 5L * DD, bo, nullptr, out + (long)t * BD);
}
